// VQVAE2_81905026335169
// MI455X (gfx1250) — matrix operand order repaired and checked by reading, not compiled, not run on hardware
//
#include <hip/hip_runtime.h>
#include <hip/hip_bf16.h>

// ---------------------------------------------------------------------------
// VQ-VAE-2 forward for MI455X (gfx1250).  All conv / convT compute is lowered
// to V_WMMA_F32_16X16X32_BF16 implicit-GEMM (bf16 in, f32 accumulate).
// Weights are pre-converted to bf16; per-chunk weight tiles are staged in LDS
// (double-buffered b64 copies) and read back as b128 fragments.
// Per-wave tile: 16 Cout x 64 pixels (4 accumulators share one A fragment).
// ---------------------------------------------------------------------------

typedef __attribute__((ext_vector_type(16))) __bf16 v16bf;
typedef __attribute__((ext_vector_type(8)))  float  v8f;

#define FUSE_RELU_OUT 1
#define FUSE_ACC      2
#define FUSE_RELU_IN  4

// GEMM view:  D[Cout, N*OH*OW] = W[Cout, Cin*KH*KW] x im2col.
// VGPR layouts used here (wave32; A/B in contiguous k order, not ISA table order):
//   A (16x32 bf16): m = lane&15, khalf = lane>>4;
//                   a[j] = A[m][16*khalf + j]
//                   -> two contiguous 8-element (16B) runs per lane
//   B (32x16 bf16): n = lane&15; b[j] = B[16*khalf + j][n]
//   C/D (16x16 f32): acc[r] = C[r + 8*khalf][n]
template <int KH, int KW>
__global__ __launch_bounds__(128) void vqvae_conv_wmma(
    const float* __restrict__ in, const __bf16* __restrict__ wgt,
    const float* __restrict__ bias, float* __restrict__ out,
    int Nimg, int Cin, int Hin, int Win, int Cout,
    int stride, int pad, int OH, int OW, int fuse)
{
    constexpr int KHW = KH * KW;
    __shared__ alignas(64) __bf16 As[2][16 * 32];   // [buf][m*32 + k]

    const int tid   = threadIdx.y * 32 + threadIdx.x;
    const int lane  = threadIdx.x;
    const int n16   = lane & 15;
    const int khalf = lane >> 4;
    const int K     = Cin * KHW;
    const int pixPerImg = OH * OW;
    const int numPix    = Nimg * pixPerImg;
    const int coutBase  = blockIdx.y << 4;
    const int waveBase  = (blockIdx.x * 4 + threadIdx.y) * 64;   // 64 px per wave

    // Per-lane B-column (output pixel) coordinates for the 4 tiles.
    int nimg[4], iy0[4], ix0[4], oy[4], ox[4];
    bool pv[4];
#pragma unroll
    for (int t = 0; t < 4; ++t) {
        int p  = waveBase + t * 16 + n16;
        pv[t]  = (p < numPix);
        int pc = pv[t] ? p : 0;
        nimg[t] = pc / pixPerImg;
        int prem = pc - nimg[t] * pixPerImg;
        oy[t] = prem / OW;
        ox[t] = prem - oy[t] * OW;
        iy0[t] = oy[t] * stride - pad;
        ix0[t] = ox[t] * stride - pad;
    }

    // Cooperative A fill: 128 threads x 4 bf16 (8B) = 16 rows x 32 k per chunk.
    // K is always a multiple of 4 here, so 4-element groups are never partial.
    const int  arow   = tid >> 3;              // 0..15
    const int  acol   = (tid & 7) * 4;         // 0,4,...,28
    const bool avalid = (coutBase + arow) < Cout;
    const __bf16* wrowA = wgt + (size_t)(avalid ? (coutBase + arow) : 0) * K;

    auto fillA = [&](int k0, int buf) {
        uint2 v = {0u, 0u};
        int ka = k0 + acol;
        if (avalid && ka + 4 <= K) v = *(const uint2*)(wrowA + ka);  // b64 load
        *(uint2*)&As[buf][arow * 32 + acol] = v;                     // ds_store_b64
    };

    v8f acc[4] = {};
    fillA(0, 0);
    __syncthreads();

    int buf = 0;
    for (int k0 = 0; k0 < K; k0 += 32, buf ^= 1) {
        __builtin_prefetch(wrowA + k0 + 64, 0, 1);   // global_prefetch_b8

        // A fragment: two 16B runs out of LDS (ds_load_b128).
        const __bf16* ar = &As[buf][n16 * 32];
        v16bf a;
        *((float4*)&a)     = *(const float4*)(ar + 16 * khalf);       // a[0..7]
        *((float4*)&a + 1) = *(const float4*)(ar + 16 * khalf + 8);   // a[8..15]

        // B fragments: kb = k0 + 16*khalf + j, j=0..15 consecutive.
        const int kb0 = k0 + 16 * khalf;
        v16bf bb[4];

        if constexpr (KHW == 16) {
            // 16-aligned fragment == exactly one ci, ky=0..3, kx=0..3.
            const int  ci     = kb0 >> 4;
            const bool kvalid = ci < Cin;
#pragma unroll
            for (int t = 0; t < 4; ++t) {
                const float* basep =
                    in + ((size_t)nimg[t] * Cin + ci) * ((size_t)Hin * Win);
                const bool tok = pv[t] && kvalid;
                const int  x0  = ix0[t];
#pragma unroll
                for (int ky = 0; ky < 4; ++ky) {
                    int iy = iy0[t] + ky;
                    bool rowok = tok && iy >= 0 && iy < Hin;
                    const float* rp = basep + (size_t)iy * Win + x0;
                    float4 r = {0.f, 0.f, 0.f, 0.f};
                    if (rowok && x0 >= 0 && x0 + 4 <= Win) {
                        r = *(const float4*)rp;                  // interior fast path
                    } else if (rowok) {
                        if (x0 + 0 >= 0 && x0 + 0 < Win) r.x = rp[0];
                        if (x0 + 1 >= 0 && x0 + 1 < Win) r.y = rp[1];
                        if (x0 + 2 >= 0 && x0 + 2 < Win) r.z = rp[2];
                        if (x0 + 3 >= 0 && x0 + 3 < Win) r.w = rp[3];
                    }
                    if (fuse & FUSE_RELU_IN) {
                        r.x = fmaxf(r.x, 0.f); r.y = fmaxf(r.y, 0.f);
                        r.z = fmaxf(r.z, 0.f); r.w = fmaxf(r.w, 0.f);
                    }
                    bb[t][ky * 4 + 0] = (__bf16)r.x;
                    bb[t][ky * 4 + 1] = (__bf16)r.y;
                    bb[t][ky * 4 + 2] = (__bf16)r.z;
                    bb[t][ky * 4 + 3] = (__bf16)r.w;
                }
            }
        } else {
            // Generic path: decompose once, advance incrementally (folds fully).
            int ci = kb0 / KHW;
            int rr = kb0 - ci * KHW;
            int ky = rr / KW;
            int kx = rr - ky * KW;
#pragma unroll
            for (int j = 0; j < 16; ++j) {
                const bool kvalid = (kb0 + j) < K;
#pragma unroll
                for (int t = 0; t < 4; ++t) {
                    float bv = 0.0f;
                    int iy = iy0[t] + ky;
                    int ix = ix0[t] + kx;
                    if (pv[t] && kvalid && iy >= 0 && iy < Hin && ix >= 0 && ix < Win) {
                        bv = in[(((size_t)nimg[t] * Cin + ci) * Hin + iy) * Win + ix];
                        if (fuse & FUSE_RELU_IN) bv = fmaxf(bv, 0.0f);
                    }
                    bb[t][j] = (__bf16)bv;
                }
                if (++kx == KW) { kx = 0; if (++ky == KH) { ky = 0; ++ci; } }
            }
        }

#pragma unroll
        for (int t = 0; t < 4; ++t)
            acc[t] = __builtin_amdgcn_wmma_f32_16x16x32_bf16(
                         false, a, false, bb[t], (short)0, acc[t], false, false);

        if (k0 + 32 < K) fillA(k0 + 32, buf ^ 1);
        __syncthreads();   // one barrier/iter is enough with double buffering
    }

    // Writeback: acc[t][r] -> cout = coutBase + r + 8*khalf, pixel tile t col n16.
#pragma unroll
    for (int t = 0; t < 4; ++t) {
        if (!pv[t]) continue;
#pragma unroll
        for (int r = 0; r < 8; ++r) {
            int co = coutBase + r + 8 * khalf;
            if (co >= Cout) continue;
            float v = acc[t][r] + bias[co];
            size_t oidx = (((size_t)nimg[t] * Cout + co) * OH + oy[t]) * OW + ox[t];
            if (fuse & FUSE_ACC)      v += out[oidx];
            if (fuse & FUSE_RELU_OUT) v  = fmaxf(v, 0.0f);
            out[oidx] = v;
        }
    }
}

// f32 -> bf16 weight conversion.
__global__ void vqvae_w2bf(const float* __restrict__ w, __bf16* __restrict__ o, int n)
{
    int i = blockIdx.x * blockDim.x + threadIdx.x;
    if (i < n) o[i] = (__bf16)w[i];
}

// Zero-stuffed dilation (lhs_dilation=2): [N,C,H,W] -> [N,C,2H-1,2W-1].
__global__ void vqvae_dilate(const float* __restrict__ in, float* __restrict__ out,
                             int N, int C, int H, int W, int relu)
{
    const int Hd = 2 * H - 1, Wd = 2 * W - 1;
    const long total = (long)N * C * Hd * Wd;
    long i = (long)blockIdx.x * blockDim.x + threadIdx.x;
    if (i >= total) return;
    int x = (int)(i % Wd); long t = i / Wd;
    int y = (int)(t % Hd); t /= Hd;
    int c = (int)(t % C);  int n = (int)(t / C);
    float v = 0.0f;
    if (((x | y) & 1) == 0) {
        v = in[(((size_t)n * C + c) * H + (y >> 1)) * W + (x >> 1)];
        if (relu) v = fmaxf(v, 0.0f);
    }
    out[i] = v;
}

// Channel-concat copy (optionally fusing a ReLU on the source).
__global__ void vqvae_concat(const float* __restrict__ src, float* __restrict__ dst,
                             int N, int Csrc, int Ctot, int coff, int HW, int relu)
{
    const long total = (long)N * Csrc * HW;
    long i = (long)blockIdx.x * blockDim.x + threadIdx.x;
    if (i >= total) return;
    int r = (int)(i % HW); long t = i / HW;
    int c = (int)(t % Csrc); int n = (int)(t / Csrc);
    float v = src[i];
    if (relu) v = fmaxf(v, 0.0f);
    dst[((size_t)n * Ctot + coff + c) * HW + r] = v;
}

// ConvTranspose weight transform: wt[o][i][y][x] = (bf16) w[i][o][3-y][3-x]  (k=4).
__global__ void vqvae_wtflip(const float* __restrict__ w, __bf16* __restrict__ wt,
                             int Cin, int Cout)
{
    int i = blockIdx.x * blockDim.x + threadIdx.x;
    int total = Cin * Cout * 16;
    if (i >= total) return;
    int kx = i & 3, ky = (i >> 2) & 3;
    int t = i >> 4; int ci = t % Cin; int co = t / Cin;
    wt[i] = (__bf16)w[(((size_t)ci * Cout + co) * 4 + (3 - ky)) * 4 + (3 - kx)];
}

// Vector quantize: per pixel, argmin over nemb 64-dim codes (NCHW strided).
// Straight-through forward value is the code itself; loss = (1+beta)*mean(d).
__global__ void vqvae_vq(const float* __restrict__ z, const float* __restrict__ emb,
                         float* __restrict__ zq, float* __restrict__ sse,
                         int N, int HW, int nemb)
{
    int p = blockIdx.x * blockDim.x + threadIdx.x;
    int total = N * HW;
    if (p >= total) return;
    int n = p / HW, r = p - n * HW;
    const float* zp = z + ((size_t)n * 64) * HW + r;
    float zv[64];
#pragma unroll
    for (int c = 0; c < 64; ++c) zv[c] = zp[(size_t)c * HW];
    int best = 0; float bestd = 3.4e38f;
    for (int e = 0; e < nemb; ++e) {
        const float* ev = emb + (size_t)e * 64;
        float d = 0.0f;
#pragma unroll
        for (int c = 0; c < 64; ++c) { float t = zv[c] - ev[c]; d = fmaf(t, t, d); }
        if (d < bestd) { bestd = d; best = e; }
    }
    const float* ev = emb + (size_t)best * 64;
    float* op = zq + ((size_t)n * 64) * HW + r;
#pragma unroll
    for (int c = 0; c < 64; ++c) op[(size_t)c * HW] = ev[c];
    atomicAdd(sse, bestd);
}

__global__ void vqvae_zero(float* p, int n)
{
    int i = blockIdx.x * blockDim.x + threadIdx.x;
    if (i < n) p[i] = 0.0f;
}

__global__ void vqvae_loss(const float* __restrict__ sse, float* __restrict__ out,
                           float inv_t, float inv_b)
{
    // diff_t + diff_b, each = beta*mean + mean = 1.25 * mean(||z_q - z||^2)
    out[0] = 1.25f * (sse[0] * inv_t + sse[1] * inv_b);
}

// ---------------------------------------------------------------------------
// Host orchestration
// ---------------------------------------------------------------------------
extern "C" void kernel_launch(void* const* d_in, const int* in_sizes, int n_in,
                              void* d_out, int out_size, void* d_ws, size_t ws_size,
                              hipStream_t stream)
{
    (void)in_sizes; (void)n_in; (void)ws_size;
    // d_in leaf order (setup_inputs dict-insertion order, {w,b} per conv):
    //  0:x  1,2:enc_b_c0  3,4:enc_b_c1  5,6:enc_b_c2  7-10:enc_b_r0  11-14:enc_b_r1
    //  15,16:enc_t_c0  17,18:enc_t_c1  19-22:enc_t_r0  23-26:enc_t_r1
    //  27,28:qconv_t  29:emb_t  30,31:dec_t_c0  32-35:dec_t_r0  36-39:dec_t_r1
    //  40,41:dec_t_ct  42,43:qconv_b  44:emb_b  45,46:up_t
    //  47,48:dec_c0  49-52:dec_r0  53-56:dec_r1  57,58:dec_ct0  59,60:dec_ct1
    auto P = [&](int i) { return (const float*)d_in[i]; };
    const float* X = P(0);

    // Workspace carve-up (float elems, 256-aligned).  Regions reused across
    // disjoint lifetimes; peak ~480 MB.
    float* ws = (float*)d_ws;
    size_t off = 0;
    auto AL = [&](size_t n) { size_t r = off; off += (n + 255) & ~(size_t)255; return ws + r; };
    float* bigA   = AL(66585600); // e0 [16,64,128,128]; later all convT dilation scratch
    float* bigB   = AL(16777216); // eb [16,192,64,64] -> qcat [16,128,64,64] -> ct0out [16,64,128,128]
    float* bufE1  = AL(8388608);  // e1 [16,128,64,64]; later dec h accumulator
    float* bufE2  = AL(8388608);  // enc_b pre-relu accumulator (persists to concat)
    float* midb   = AL(2097152);  // residual hidden [<=16,32,64,64]
    float* topA   = AL(2097152);  // enc-top accumulator; later dec_t accumulator [16,128,32,32]
    float* topC   = AL(1048576);  // t0 [16,64,32,32]; later qt
    float* quantT = AL(1048576);  // quant_t [16,64,32,32]
    float* decT   = AL(4194304);  // dec_t output [16,64,64,64]
    float* qbBuf  = AL(4194304);  // qb [16,64,64,64]; later ut
    float* quantB = AL(4194304);  // quant_b [16,64,64,64]
    __bf16* wbf   = (__bf16*)AL(73984);  // bf16 weights (max 128*128*9 = 147456 elems)
    float* lossB  = AL(256);      // [0]=sse_t, [1]=sse_b

    auto EW = [&](long n) { return dim3((unsigned)((n + 255) / 256)); };

    // Launch conv with already-bf16 weights.
    auto CONVB = [&](const float* in, const __bf16* w, const float* b, float* out,
                     int N, int Cin, int Hin, int Win, int Cout, int KH, int KW,
                     int st, int pad, int fuse) {
        int OH = (Hin + 2 * pad - KH) / st + 1;
        int OW = (Win + 2 * pad - KW) / st + 1;
        int numPix = N * OH * OW;
        dim3 g((numPix + 255) / 256, (Cout + 15) / 16), blk(32, 4, 1);
        if (KH == 4)
            vqvae_conv_wmma<4, 4><<<g, blk, 0, stream>>>(in, w, b, out, N, Cin, Hin, Win,
                                                         Cout, st, pad, OH, OW, fuse);
        else if (KH == 3)
            vqvae_conv_wmma<3, 3><<<g, blk, 0, stream>>>(in, w, b, out, N, Cin, Hin, Win,
                                                         Cout, st, pad, OH, OW, fuse);
        else
            vqvae_conv_wmma<1, 1><<<g, blk, 0, stream>>>(in, w, b, out, N, Cin, Hin, Win,
                                                         Cout, st, pad, OH, OW, fuse);
    };
    // Conv with f32 weights: convert to bf16 scratch first.
    auto CONV = [&](const float* in, const float* w, const float* b, float* out,
                    int N, int Cin, int Hin, int Win, int Cout, int KH, int KW,
                    int st, int pad, int fuse) {
        int wn = Cout * Cin * KH * KW;
        vqvae_w2bf<<<EW(wn), 256, 0, stream>>>(w, wbf, wn);
        CONVB(in, wbf, b, out, N, Cin, Hin, Win, Cout, KH, KW, st, pad, fuse);
    };
    // ConvT(k=4,s=2,p=1) == dilate(2) + pad2 + conv(flipped, IO-swapped weights).
    auto CONVT = [&](const float* in, const float* w, const float* b, float* out,
                     int N, int Cin, int Hin, int Win, int Cout, int relu_in, int fuse_out) {
        vqvae_wtflip<<<EW((long)Cin * Cout * 16), 256, 0, stream>>>(w, wbf, Cin, Cout);
        int Hd = 2 * Hin - 1, Wd = 2 * Win - 1;
        vqvae_dilate<<<EW((long)N * Cin * Hd * Wd), 256, 0, stream>>>(in, bigA, N, Cin, Hin, Win, relu_in);
        CONVB(bigA, wbf, b, out, N, Cin, Hd, Wd, Cout, 4, 4, 1, 2, fuse_out);
    };
    auto RES = [&](float* xbuf, int N, int C, int H, int W, int pi) {
        CONV(xbuf, P(pi), P(pi + 1), midb, N, C, H, W, 32, 3, 3, 1, 1, FUSE_RELU_IN | FUSE_RELU_OUT);
        CONV(midb, P(pi + 2), P(pi + 3), xbuf, N, 32, H, W, C, 1, 1, 1, 0, FUSE_ACC);
    };

    vqvae_zero<<<1, 32, 0, stream>>>(lossB, 2);

    // -------- Encoder bottom (stride 4) --------
    CONV(X, P(1), P(2), bigA, 16, 3, 256, 256, 64, 4, 4, 2, 1, FUSE_RELU_OUT);       // e0
    CONV(bigA, P(3), P(4), bufE1, 16, 64, 128, 128, 128, 4, 4, 2, 1, FUSE_RELU_OUT); // e1
    CONV(bufE1, P(5), P(6), bufE2, 16, 128, 64, 64, 128, 3, 3, 1, 1, 0);             // e2
    RES(bufE2, 16, 128, 64, 64, 7);
    RES(bufE2, 16, 128, 64, 64, 11);
    // enc_b = relu(bufE2)  (fused into downstream consumers)

    // -------- Encoder top (stride 2) --------
    CONV(bufE2, P(15), P(16), topC, 16, 128, 64, 64, 64, 4, 4, 2, 1, FUSE_RELU_IN | FUSE_RELU_OUT);
    CONV(topC, P(17), P(18), topA, 16, 64, 32, 32, 128, 3, 3, 1, 1, 0);
    RES(topA, 16, 128, 32, 32, 19);
    RES(topA, 16, 128, 32, 32, 23);

    // -------- Top quantize --------
    CONV(topA, P(27), P(28), topC, 16, 128, 32, 32, 64, 1, 1, 1, 0, FUSE_RELU_IN);   // qt
    vqvae_vq<<<EW(16 * 1024), 256, 0, stream>>>(topC, P(29), quantT, lossB + 0, 16, 1024, 512);

    // -------- dec_t: quant_t -> [16,64,64,64] --------
    CONV(quantT, P(30), P(31), topA, 16, 64, 32, 32, 128, 3, 3, 1, 1, 0);
    RES(topA, 16, 128, 32, 32, 32);
    RES(topA, 16, 128, 32, 32, 36);
    CONVT(topA, P(40), P(41), decT, 16, 128, 32, 32, 64, /*relu_in=*/1, 0);

    // -------- Bottom quantize --------
    vqvae_concat<<<EW((long)16 * 64 * 4096), 256, 0, stream>>>(decT, bigB, 16, 64, 192, 0, 4096, 0);
    vqvae_concat<<<EW((long)16 * 128 * 4096), 256, 0, stream>>>(bufE2, bigB, 16, 128, 192, 64, 4096, 1);
    CONV(bigB, P(42), P(43), qbBuf, 16, 192, 64, 64, 64, 1, 1, 1, 0, 0);             // qb
    vqvae_vq<<<EW(16 * 4096), 256, 0, stream>>>(qbBuf, P(44), quantB, lossB + 1, 16, 4096, 512);

    // -------- Decode (stride 4) --------
    CONVT(quantT, P(45), P(46), qbBuf, 16, 64, 32, 32, 64, 0, 0);                    // ut (reuse qbBuf)
    vqvae_concat<<<EW((long)16 * 64 * 4096), 256, 0, stream>>>(qbBuf, bigB, 16, 64, 128, 0, 4096, 0);
    vqvae_concat<<<EW((long)16 * 64 * 4096), 256, 0, stream>>>(quantB, bigB, 16, 64, 128, 64, 4096, 0);
    CONV(bigB, P(47), P(48), bufE1, 16, 128, 64, 64, 128, 3, 3, 1, 1, 0);
    RES(bufE1, 16, 128, 64, 64, 49);
    RES(bufE1, 16, 128, 64, 64, 53);
    CONVT(bufE1, P(57), P(58), bigB, 16, 128, 64, 64, 64, /*relu_in=*/1, FUSE_RELU_OUT);
    CONVT(bigB, P(59), P(60), (float*)d_out, 16, 64, 128, 128, 30, 0, 0);            // dec

    // -------- latent_loss --------
    vqvae_loss<<<1, 1, 0, stream>>>(lossB, (float*)d_out + (out_size - 1),
                                    1.0f / (16.0f * 1024.0f * 64.0f),
                                    1.0f / (16.0f * 4096.0f * 64.0f));
}
